// StaticDictionary_79998060855635
// MI455X (gfx1250) — compile-verified, matrix operand order checked by reading, not run on hardware
//
#include <hip/hip_runtime.h>
#include <hip/hip_bf16.h>

typedef __bf16 v16bf __attribute__((ext_vector_type(16)));
typedef __bf16 v8bf  __attribute__((ext_vector_type(8)));
typedef float  v8f   __attribute__((ext_vector_type(8)));

#define NKEYS   200000
#define NQ      1024
#define DIM     128
#define KNN     50
#define NCHUNK  25
#define CHUNK   8000            // 25 * 8000 = 200000, 8000 = 125 * 64
#define STEPS   125             // per workgroup: 125 steps of 64 keys (4 waves * 16)
#define CAND    (NCHUNK * KNN)  // 1250 candidates per query
#define BIGF    3.0e38f

// ---------------------------------------------------------------------------
// Phase 0: one pass over rows of 128 fp32:
//   - split each element into hi=bf16(x), lo=bf16(x-hi)  (3xbf16 ~ fp32 trick)
//   - compute the row squared norm
// Keys are read once from HBM; the bf16 copies (102 MB total) stay L2-resident
// for the 64 query-tile re-reads in phase 1.
// ---------------------------------------------------------------------------
__global__ __launch_bounds__(128) void cvt_rows_kernel(
    const float* __restrict__ src, __bf16* __restrict__ hi,
    __bf16* __restrict__ lo, float* __restrict__ nrm, int rows) {
  int r = blockIdx.x * blockDim.x + threadIdx.x;
  if (r >= rows) return;
  const float4* p = reinterpret_cast<const float4*>(src + (size_t)r * DIM);
  const size_t base = (size_t)r * DIM;
  float s = 0.f;
#pragma unroll
  for (int i = 0; i < DIM / 8; ++i) {
    float4 a = p[2 * i], b = p[2 * i + 1];
    float f[8] = {a.x, a.y, a.z, a.w, b.x, b.y, b.z, b.w};
    v8bf h, l;
#pragma unroll
    for (int e = 0; e < 8; ++e) {
      s += f[e] * f[e];
      __bf16 hb = (__bf16)f[e];
      h[e] = hb;
      l[e] = (__bf16)(f[e] - (float)hb);
    }
    *reinterpret_cast<v8bf*>(hi + base + i * 8) = h;
    *reinterpret_cast<v8bf*>(lo + base + i * 8) = l;
  }
  nrm[r] = s;
}

__device__ inline v8f wmma_bf16(v16bf a, v16bf b, v8f c) {
  // D = A(16x32) * B(32x16) + C, fp32 accumulate
  return __builtin_amdgcn_wmma_f32_16x16x32_bf16(false, a, false, b, (short)0, c,
                                                 false, false);
}

// ---------------------------------------------------------------------------
// Phase 1: split-bf16 WMMA distance GEMM + per-chunk top-50 selection
// grid = (64 query tiles, 25 key chunks); block = 128 threads = 4 waves
// Inner tile: 16 b128-class loads + 12 v_wmma_f32_16x16x32_bf16, no conversions.
// ---------------------------------------------------------------------------
__global__ __launch_bounds__(128) void knn_gemm_select(
    const __bf16* __restrict__ q_hi, const __bf16* __restrict__ q_lo,
    const __bf16* __restrict__ k_hi, const __bf16* __restrict__ k_lo,
    const float* __restrict__ qsq, const float* __restrict__ dsq,
    float* __restrict__ cand_d, int* __restrict__ cand_i) {
  __shared__ float tileD[4][16][16];
  __shared__ float listD[4][16][KNN];
  __shared__ int   listI[4][16][KNN];
  __shared__ float thr[4][16];
  __shared__ int   pos[4][16];

  const int tid   = threadIdx.x;
  const int wave  = tid >> 5;
  const int lane  = tid & 31;
  const int mrow  = lane & 15;     // A: query row / B: key row (column of B)
  const int khalf = lane >> 4;     // which 16-wide K slice this lane holds
  const int qt    = blockIdx.x;
  const int chunk = blockIdx.y;

  // init selection state
  for (int t = tid; t < 4 * 16 * KNN; t += 128) {
    (&listD[0][0][0])[t] = BIGF;
    (&listI[0][0][0])[t] = 0;
  }
  if (tid < 64) { (&thr[0][0])[tid] = BIGF; (&pos[0][0])[tid] = 0; }

  // A fragments: 16 queries x 128 K, hi/lo bf16, resident in VGPRs
  const __bf16* qh = q_hi + (size_t)(qt * 16 + mrow) * DIM;
  const __bf16* ql = q_lo + (size_t)(qt * 16 + mrow) * DIM;
  v16bf ah[4], al[4];
#pragma unroll
  for (int ks = 0; ks < 4; ++ks) {
    ah[ks] = *reinterpret_cast<const v16bf*>(qh + ks * 32 + khalf * 16);
    al[ks] = *reinterpret_cast<const v16bf*>(ql + ks * 32 + khalf * 16);
  }

  // per-lane query norms for the 8 output rows this lane's half covers
  float qs[8];
#pragma unroll
  for (int r = 0; r < 8; ++r) qs[r] = qsq[qt * 16 + r + 8 * khalf];

  __syncthreads();

  const int chunkBase = chunk * CHUNK;
  for (int step = 0; step < STEPS; ++step) {
    const int n0 = chunkBase + step * 64 + wave * 16;  // wave-uniform
    const __bf16* kh = k_hi + (size_t)(n0 + mrow) * DIM;
    const __bf16* kl = k_lo + (size_t)(n0 + mrow) * DIM;

    if (step + 1 < STEPS) {  // prefetch next tile's key rows (L2-resident)
      __builtin_prefetch(kh + (size_t)64 * DIM + khalf * 64, 0, 3);
      __builtin_prefetch(kl + (size_t)64 * DIM + khalf * 64, 0, 3);
    }

    v8f acc = {};
#pragma unroll
    for (int ks = 0; ks < 4; ++ks) {
      v16bf bh = *reinterpret_cast<const v16bf*>(kh + ks * 32 + khalf * 16);
      v16bf bl = *reinterpret_cast<const v16bf*>(kl + ks * 32 + khalf * 16);
      acc = wmma_bf16(ah[ks], bh, acc);  // hi*hi
      acc = wmma_bf16(ah[ks], bl, acc);  // hi*lo
      acc = wmma_bf16(al[ks], bh, acc);  // lo*hi
    }

    // dist = |q|^2 + |k|^2 - 2 q.k ; stage to LDS for the selection pass
    const float dv = dsq[n0 + mrow];     // column n = n0 + lane%16
#pragma unroll
    for (int r = 0; r < 8; ++r)
      tileD[wave][r + 8 * khalf][mrow] = qs[r] + dv - 2.0f * acc[r];

    __syncthreads();

    // lanes 0..15 of each wave maintain that wave's private top-50 lists
    if (lane < 16) {
      float t = thr[wave][lane];
      int   p = pos[wave][lane];
#pragma unroll 1
      for (int c = 0; c < 16; ++c) {
        float d = tileD[wave][lane][c];
        if (d < t) {
          listD[wave][lane][p] = d;
          listI[wave][lane][p] = n0 + c;
          float mx = -BIGF; int mp = 0;
#pragma unroll 1
          for (int j = 0; j < KNN; ++j) {
            float lv = listD[wave][lane][j];
            if (lv > mx) { mx = lv; mp = j; }
          }
          t = mx; p = mp;
        }
      }
      thr[wave][lane] = t;
      pos[wave][lane] = p;
    }
    __syncthreads();
  }

  // merge wave lists 1..3 into wave 0's list (one maintainer lane per query)
  if (tid < 16) {
    float t = thr[0][tid];
    int   p = pos[0][tid];
#pragma unroll 1
    for (int src = 1; src < 4; ++src) {
#pragma unroll 1
      for (int j = 0; j < KNN; ++j) {
        float d = listD[src][tid][j];
        if (d < t) {
          listD[0][tid][p] = d;
          listI[0][tid][p] = listI[src][tid][j];
          float mx = -BIGF; int mp = 0;
#pragma unroll 1
          for (int k = 0; k < KNN; ++k) {
            float lv = listD[0][tid][k];
            if (lv > mx) { mx = lv; mp = k; }
          }
          t = mx; p = mp;
        }
      }
    }
  }
  __syncthreads();

  // emit 50 candidates per query for this chunk
  for (int t = tid; t < 16 * KNN; t += 128) {
    int qq = t / KNN, j = t % KNN;
    size_t o = ((size_t)(qt * 16 + qq) * NCHUNK + chunk) * KNN + j;
    cand_d[o] = listD[0][qq][j];
    cand_i[o] = listI[0][qq][j];
  }
}

// ---------------------------------------------------------------------------
// Phase 2: per-query merge of 1250 candidates -> top-50 -> IDW output
// one wave (32 threads) per query
// ---------------------------------------------------------------------------
__global__ __launch_bounds__(32) void knn_merge_idw(
    const float* __restrict__ cand_d, const int* __restrict__ cand_i,
    const float* __restrict__ values, float* __restrict__ out) {
  __shared__ float ld[CAND];
  __shared__ int   li[CAND];
  const int q = blockIdx.x;
  const int lane = threadIdx.x;

  for (int t = lane; t < CAND; t += 32) {
    ld[t] = cand_d[(size_t)q * CAND + t];
    li[t] = cand_i[(size_t)q * CAND + t];
  }
  __syncthreads();

  float wsum = 0.f, wvsum = 0.f, d0 = 0.f, v0 = 0.f;
#pragma unroll 1
  for (int it = 0; it < KNN; ++it) {
    float bd = BIGF; int bt = 0;
    for (int t = lane; t < CAND; t += 32) {
      float d = ld[t];
      if (d < bd) { bd = d; bt = t; }
    }
#pragma unroll
    for (int off = 16; off > 0; off >>= 1) {
      float od = __shfl_down(bd, off);
      int   ot = __shfl_down(bt, off);
      if (od < bd) { bd = od; bt = ot; }
    }
    if (lane == 0) {
      float d = bd < 0.f ? 0.f : bd;  // clamp fp error, like reference
      float v = values[li[bt]];
      if (it == 0) { d0 = d; v0 = v; }
      float w = 1.0f / (d + 1e-3f);
      wsum += w; wvsum += w * v;
      ld[bt] = BIGF;                  // remove from pool
    }
    __syncthreads();
  }
  if (lane == 0) out[q] = (d0 == 0.0f) ? v0 : (wvsum / wsum);
}

// ---------------------------------------------------------------------------
extern "C" void kernel_launch(void* const* d_in, const int* in_sizes, int n_in,
                              void* d_out, int out_size, void* d_ws, size_t ws_size,
                              hipStream_t stream) {
  const float* qk     = (const float*)d_in[0];  // [1024,128]
  const float* keys   = (const float*)d_in[1];  // [200000,128]
  const float* values = (const float*)d_in[2];  // [200000,1]
  float* out = (float*)d_out;                   // [1024]

  char* ws = (char*)d_ws;
  __bf16* k_hi  = (__bf16*)ws;  ws += (size_t)NKEYS * DIM * 2;  // 51.2 MB
  __bf16* k_lo  = (__bf16*)ws;  ws += (size_t)NKEYS * DIM * 2;  // 51.2 MB
  __bf16* q_hi  = (__bf16*)ws;  ws += (size_t)NQ * DIM * 2;     // 256 KB
  __bf16* q_lo  = (__bf16*)ws;  ws += (size_t)NQ * DIM * 2;     // 256 KB
  float*  dsq   = (float*)ws;   ws += (size_t)NKEYS * 4;        // 800 KB
  float*  qsq   = (float*)ws;   ws += (size_t)NQ * 4;           // 4 KB
  float*  cand_d = (float*)ws;  ws += (size_t)NQ * CAND * 4;    // 5.12 MB
  int*    cand_i = (int*)ws;

  cvt_rows_kernel<<<(NKEYS + 127) / 128, 128, 0, stream>>>(keys, k_hi, k_lo, dsq, NKEYS);
  cvt_rows_kernel<<<(NQ + 127) / 128, 128, 0, stream>>>(qk, q_hi, q_lo, qsq, NQ);

  dim3 g1(NQ / 16, NCHUNK, 1);
  knn_gemm_select<<<g1, 128, 0, stream>>>(q_hi, q_lo, k_hi, k_lo, qsq, dsq,
                                          cand_d, cand_i);

  knn_merge_idw<<<NQ, 32, 0, stream>>>(cand_d, cand_i, values, out);
}
